// PoolHiddenNet_5360119185918
// MI455X (gfx1250) — compile-verified, matrix operand order checked by reading, not run on hardware
//
#include <hip/hip_runtime.h>
#include <hip/hip_bf16.h>

// PoolHiddenNet on MI455X (gfx1250, wave32, WMMA).
// Compute-bound: ~155 GFLOP of GEMM vs ~300MB of HBM traffic -> f16 WMMA
// (v_wmma_f32_16x16x32_f16) with f32 accumulation, fused BN+ReLU(+maxpool)
// epilogues, double-buffered LDS staging, and CDNA5 async global->LDS
// copies (global_load_async_to_lds_b128, tracked with ASYNCcnt).

typedef __attribute__((ext_vector_type(16))) _Float16 v16h;
typedef __attribute__((ext_vector_type(8)))  _Float16 v8h;
typedef __attribute__((ext_vector_type(8)))  float    v8f;
typedef __attribute__((ext_vector_type(4)))  int      v4i;

#define S_    32
#define N_    16
#define ROWS  256     // N*N rows per scene
#define KIN   128     // E + H
#define D1_   8192
#define BD_   1024

// ------------------------------------------------ async global->LDS path --
#if __has_builtin(__builtin_amdgcn_global_load_async_to_lds_b128)
#define HAVE_ASYNC_LDS 1
#else
#define HAVE_ASYNC_LDS 0
#endif

__device__ inline void copy16_g2l(void* lds, const void* g)
{
#if HAVE_ASYNC_LDS
    // global_load_async_to_lds_b128: bypasses VGPRs, tracked by ASYNCcnt.
    // Param 0 is the global (AS1) source, param 1 the LDS (AS3) dest.
    __builtin_amdgcn_global_load_async_to_lds_b128(
        (__attribute__((address_space(1))) v4i*)g,
        (__attribute__((address_space(3))) v4i*)lds,
        0, 0);
#else
    *(uint4*)lds = *(const uint4*)g;       // global_load_b128 + ds_store_b128
#endif
}

__device__ inline void g2l_wait_all()
{
#if HAVE_ASYNC_LDS
#if __has_builtin(__builtin_amdgcn_s_wait_asynccnt)
    __builtin_amdgcn_s_wait_asynccnt(0);
#else
    asm volatile("s_wait_asynccnt 0x0" ::: "memory");
#endif
#endif
    // sync fallback: __syncthreads() below already waits dscnt/loadcnt
}

// ---------------------------------------------------------------- prep ----
__global__ __launch_bounds__(256) void build_x_kernel(
    const float* __restrict__ h, const float* __restrict__ pos,
    const float* __restrict__ Wsp, const float* __restrict__ bsp,
    _Float16* __restrict__ Xh)
{
    int tid = blockIdx.x * 256 + threadIdx.x;        // S*ROWS*KIN = 1M
    int r = tid >> 7, c = tid & 127;
    int s = r >> 8, rem = r & 255, i = rem >> 4, j = rem & 15;
    int pi = s * N_ + i, pj = s * N_ + j;
    float val;
    if (c < 64) {
        // rel = clip(pos[j]-pos[i], -NS/2, NS/2)/(NS/2); NS/2 == 1
        float rx = fminf(fmaxf(pos[pj*2+0] - pos[pi*2+0], -1.f), 1.f);
        float ry = fminf(fmaxf(pos[pj*2+1] - pos[pi*2+1], -1.f), 1.f);
        val = rx * Wsp[c] + ry * Wsp[64 + c] + bsp[c];
    } else {
        val = h[pj * 64 + (c - 64)];
    }
    Xh[tid] = (_Float16)val;
}

__global__ __launch_bounds__(256) void conv_w1_kernel(
    const float* __restrict__ W, _Float16* __restrict__ Wt)
{
    int tid = blockIdx.x * 256 + threadIdx.x;        // D1*KIN
    int c = tid >> 7, k = tid & 127;                 // Wt[c][k] = W[k][c]
    Wt[tid] = (_Float16)W[k * D1_ + c];
}

__global__ __launch_bounds__(256) void conv_w2_kernel(
    const float* __restrict__ W, _Float16* __restrict__ Wt)
{
    size_t tid = (size_t)blockIdx.x * 256 + threadIdx.x;  // BD*D1 = 8M
    size_t c = tid >> 13, k = tid & 8191;            // Wt[c][k] = W[k][c]
    Wt[tid] = (_Float16)W[k * BD_ + c];
}

// ------------------------------------------------------- WMMA fragments ---
// 16-bit A (16x32, MxK) / B (32x16 col-per-lane) fragment from an LDS tile
// stored as [16 rows][32 k] f16 (K contiguous). ISA layout: lane<16 -> K
// 0..7,16..23 ; lane>=16 -> K 8..15,24..31 ; K pairs packed per VGPR.
__device__ inline v16h load_frag_lds(const _Float16* tile, int lane)
{
    int m = lane & 15, kh = lane >> 4;
    const _Float16* rp = tile + m * 32;
    v8h lo = *(const v8h*)(rp + kh * 8);         // ds_load_b128
    v8h hi = *(const v8h*)(rp + 16 + kh * 8);    // ds_load_b128
    v16h f;
#pragma unroll
    for (int e = 0; e < 8; ++e) { f[e] = lo[e]; f[e + 8] = hi[e]; }
    return f;
}

// ----------------------------------------------------- layer 1 (fused) ----
// grid (D1/64, S); block 256 = 8 wave32. Wave w: row tiles 2w,2w+1; 4 col
// tiles of 16. Output: relu(BN(x@W1)) -> Y1 as f16 [S][ROWS][D1].
__global__ __launch_bounds__(256) void layer1_kernel(
    const _Float16* __restrict__ Xh, const _Float16* __restrict__ W1t,
    const float* __restrict__ g1, const float* __restrict__ be1,
    _Float16* __restrict__ Y1h)
{
    __shared__ alignas(16) _Float16 As[2][ROWS * 32];   // double buffered
    __shared__ alignas(16) _Float16 Bs[2][64 * 32];
    __shared__ float s_sum[64];
    __shared__ float s_ssq[64];

    const int cb   = blockIdx.x;      // 64-channel block
    const int s    = blockIdx.y;      // scene
    const int tid  = threadIdx.x;
    const int lane = tid & 31;
    const int wave = tid >> 5;
    const int nloc = lane & 15;
    const int cc   = tid >> 2, part = tid & 3;

    if (tid < 64) { s_sum[tid] = 0.f; s_ssq[tid] = 0.f; }

    v8f acc[2][4] = {};

    const _Float16* Xs = Xh + (size_t)s * ROWS * KIN;
    const _Float16* Bg = W1t + (size_t)(cb * 64 + cc) * KIN + part * 8;
    const int NK = KIN / 32;

    // prologue: chunk 0 -> buffer 0
    {
        const _Float16* srcA = Xs + (size_t)tid * KIN;
        copy16_g2l(&As[0][tid * 32 +  0], srcA +  0);
        copy16_g2l(&As[0][tid * 32 +  8], srcA +  8);
        copy16_g2l(&As[0][tid * 32 + 16], srcA + 16);
        copy16_g2l(&As[0][tid * 32 + 24], srcA + 24);
        copy16_g2l(&Bs[0][cc * 32 + part * 8], Bg);
    }

    for (int kt = 0; kt < NK; ++kt) {
        const int cur = kt & 1;
        g2l_wait_all();
        __syncthreads();                 // chunk kt resident for all waves
        if (kt + 1 < NK) {               // overlap: fetch kt+1 into other buf
            const int nxt = cur ^ 1;
            const _Float16* srcA = Xs + (size_t)tid * KIN + (kt + 1) * 32;
            copy16_g2l(&As[nxt][tid * 32 +  0], srcA +  0);
            copy16_g2l(&As[nxt][tid * 32 +  8], srcA +  8);
            copy16_g2l(&As[nxt][tid * 32 + 16], srcA + 16);
            copy16_g2l(&As[nxt][tid * 32 + 24], srcA + 24);
            copy16_g2l(&Bs[nxt][cc * 32 + part * 8], Bg + (kt + 1) * 32);
        }
        v16h bfrag[4];
#pragma unroll
        for (int ct = 0; ct < 4; ++ct)
            bfrag[ct] = load_frag_lds(&Bs[cur][ct * 16 * 32], lane);
#pragma unroll
        for (int ti = 0; ti < 2; ++ti) {
            int t = wave * 2 + ti;
            v16h afrag = load_frag_lds(&As[cur][t * 16 * 32], lane);
#pragma unroll
            for (int ct = 0; ct < 4; ++ct)
                acc[ti][ct] = __builtin_amdgcn_wmma_f32_16x16x32_f16(
                    false, afrag, false, bfrag[ct],
                    (short)0, acc[ti][ct], false, false);
        }
    }
    __syncthreads();

    // BN stats across all 256 rows (b1 cancels under BN -> skipped)
#pragma unroll
    for (int ti = 0; ti < 2; ++ti)
#pragma unroll
        for (int ct = 0; ct < 4; ++ct) {
            float sum = 0.f, ssq = 0.f;
#pragma unroll
            for (int v = 0; v < 8; ++v) { float x = acc[ti][ct][v]; sum += x; ssq += x * x; }
            atomicAdd(&s_sum[ct * 16 + nloc], sum);   // ds_add_f32
            atomicAdd(&s_ssq[ct * 16 + nloc], ssq);
        }
    __syncthreads();

    const float inv_n = 1.0f / 256.0f;
#pragma unroll
    for (int ti = 0; ti < 2; ++ti) {
        int t = wave * 2 + ti;
#pragma unroll
        for (int ct = 0; ct < 4; ++ct) {
            int c  = ct * 16 + nloc;
            int gc = cb * 64 + c;
            float mean = s_sum[c] * inv_n;
            float var  = s_ssq[c] * inv_n - mean * mean;
            float inv  = rsqrtf(var + 1e-5f);
            float ga = g1[gc] * inv;
            float bb = be1[gc] - mean * ga;
#pragma unroll
            for (int v = 0; v < 8; ++v) {
                int m   = v + (lane >> 4) * 8;
                int row = t * 16 + m;
                float y = acc[ti][ct][v] * ga + bb;
                y = y > 0.f ? y : 0.f;
                Y1h[((size_t)s * ROWS + row) * D1_ + gc] = (_Float16)y;
            }
        }
    }
}

// ---------------------------------------------- layer 2 + maxpool (fused) -
// grid (BD/64, S). K = 8192 streamed in double-buffered 32-chunks. Row-tile
// t == pedestrian i, so pooling = per-tile max over 8 VGPRs + __shfl_xor.
__global__ __launch_bounds__(256) void layer2_kernel(
    const _Float16* __restrict__ Y1h, const _Float16* __restrict__ W2t,
    const float* __restrict__ g2, const float* __restrict__ be2,
    float* __restrict__ out)
{
    __shared__ alignas(16) _Float16 As[2][ROWS * 32];
    __shared__ alignas(16) _Float16 Bs[2][64 * 32];
    __shared__ float s_sum[64];
    __shared__ float s_ssq[64];

    const int cb   = blockIdx.x;
    const int s    = blockIdx.y;
    const int tid  = threadIdx.x;
    const int lane = tid & 31;
    const int wave = tid >> 5;
    const int nloc = lane & 15;
    const int cc   = tid >> 2, part = tid & 3;

    if (tid < 64) { s_sum[tid] = 0.f; s_ssq[tid] = 0.f; }

    v8f acc[2][4] = {};

    const _Float16* Ys = Y1h + (size_t)s * ROWS * D1_ + (size_t)tid * D1_;
    const _Float16* Bg = W2t + (size_t)(cb * 64 + cc) * D1_ + part * 8;
    const int NK = D1_ / 32;

    {
        copy16_g2l(&As[0][tid * 32 +  0], Ys +  0);
        copy16_g2l(&As[0][tid * 32 +  8], Ys +  8);
        copy16_g2l(&As[0][tid * 32 + 16], Ys + 16);
        copy16_g2l(&As[0][tid * 32 + 24], Ys + 24);
        copy16_g2l(&Bs[0][cc * 32 + part * 8], Bg);
    }

    for (int kt = 0; kt < NK; ++kt) {
        const int cur = kt & 1;
        g2l_wait_all();
        __syncthreads();
        if (kt + 1 < NK) {
            const int nxt = cur ^ 1;
            const _Float16* srcA = Ys + (kt + 1) * 32;
            copy16_g2l(&As[nxt][tid * 32 +  0], srcA +  0);
            copy16_g2l(&As[nxt][tid * 32 +  8], srcA +  8);
            copy16_g2l(&As[nxt][tid * 32 + 16], srcA + 16);
            copy16_g2l(&As[nxt][tid * 32 + 24], srcA + 24);
            copy16_g2l(&Bs[nxt][cc * 32 + part * 8], Bg + (kt + 1) * 32);
            if (kt + 2 < NK)
                __builtin_prefetch(srcA + 32, 0, 1);   // global_prefetch_b8
        }
        v16h bfrag[4];
#pragma unroll
        for (int ct = 0; ct < 4; ++ct)
            bfrag[ct] = load_frag_lds(&Bs[cur][ct * 16 * 32], lane);
#pragma unroll
        for (int ti = 0; ti < 2; ++ti) {
            int t = wave * 2 + ti;
            v16h afrag = load_frag_lds(&As[cur][t * 16 * 32], lane);
#pragma unroll
            for (int ct = 0; ct < 4; ++ct)
                acc[ti][ct] = __builtin_amdgcn_wmma_f32_16x16x32_f16(
                    false, afrag, false, bfrag[ct],
                    (short)0, acc[ti][ct], false, false);
        }
    }
    __syncthreads();

#pragma unroll
    for (int ti = 0; ti < 2; ++ti)
#pragma unroll
        for (int ct = 0; ct < 4; ++ct) {
            float sum = 0.f, ssq = 0.f;
#pragma unroll
            for (int v = 0; v < 8; ++v) { float x = acc[ti][ct][v]; sum += x; ssq += x * x; }
            atomicAdd(&s_sum[ct * 16 + nloc], sum);
            atomicAdd(&s_ssq[ct * 16 + nloc], ssq);
        }
    __syncthreads();

    const float inv_n = 1.0f / 256.0f;
#pragma unroll
    for (int ti = 0; ti < 2; ++ti) {
        int t = wave * 2 + ti;                        // pedestrian i
#pragma unroll
        for (int ct = 0; ct < 4; ++ct) {
            int c  = ct * 16 + nloc;
            int gc = cb * 64 + c;
            float mean = s_sum[c] * inv_n;
            float var  = s_ssq[c] * inv_n - mean * mean;
            float inv  = rsqrtf(var + 1e-5f);
            float ga = g2[gc] * inv;
            float bb = be2[gc] - mean * ga;
            float mx = 0.f;                           // relu floor
#pragma unroll
            for (int v = 0; v < 8; ++v) {
                float y = acc[ti][ct][v] * ga + bb;
                mx = fmaxf(mx, y);                    // relu(max) == max(relu)
            }
            mx = fmaxf(mx, __shfl_xor(mx, 16));       // join M halves
            if (lane < 16)
                out[((size_t)s * N_ + t) * BD_ + gc] = mx;
        }
    }
}

// ---------------------------------------------------------------- launch --
extern "C" void kernel_launch(void* const* d_in, const int* in_sizes, int n_in,
                              void* d_out, int out_size, void* d_ws, size_t ws_size,
                              hipStream_t stream) {
    const float* h_states = (const float*)d_in[0];   // (1,512,64)
    const float* end_pos  = (const float*)d_in[1];   // (512,2)
    // d_in[2] rel_pos: unused by reference
    const float* W_sp = (const float*)d_in[3];       // (2,64)
    const float* b_sp = (const float*)d_in[4];       // (64)
    const float* W1   = (const float*)d_in[5];       // (128,8192)
    // d_in[6] b1: cancels under BN
    const float* g1   = (const float*)d_in[7];
    const float* be1  = (const float*)d_in[8];
    const float* W2   = (const float*)d_in[9];       // (8192,1024)
    // d_in[10] b2: cancels under BN
    const float* g2   = (const float*)d_in[11];
    const float* be2  = (const float*)d_in[12];
    // d_in[13] seq_start_end: uniform groups of 16, implied by indexing
    float* out = (float*)d_out;                      // (512,1024) f32

    char* ws = (char*)d_ws;
    _Float16* Xh  = (_Float16*)(ws);                          //  2 MiB
    _Float16* W1t = (_Float16*)(ws + ((size_t)2  << 20));     //  2 MiB
    _Float16* W2t = (_Float16*)(ws + ((size_t)4  << 20));     // 16 MiB
    _Float16* Y1h = (_Float16*)(ws + ((size_t)20 << 20));     // 128 MiB

    build_x_kernel<<<(S_ * ROWS * KIN) / 256, 256, 0, stream>>>(
        h_states, end_pos, W_sp, b_sp, Xh);
    conv_w1_kernel<<<(D1_ * KIN) / 256, 256, 0, stream>>>(W1, W1t);
    conv_w2_kernel<<<(BD_ * D1_) / 256, 256, 0, stream>>>(W2, W2t);
    layer1_kernel<<<dim3(D1_ / 64, S_), 256, 0, stream>>>(Xh, W1t, g1, be1, Y1h);
    layer2_kernel<<<dim3(BD_ / 64, S_), 256, 0, stream>>>(Y1h, W2t, g2, be2, out);
}
